// ConformerHybrid_88141318848850
// MI455X (gfx1250) — matrix operand order repaired and checked by reading, not compiled, not run on hardware
//
#include <hip/hip_runtime.h>

#define GEPS 1e-3f

typedef unsigned short u16;
typedef unsigned int u32;
typedef __bf16 bf16_t;
typedef bf16_t bf16x16 __attribute__((ext_vector_type(16)));
typedef bf16_t bf16x8  __attribute__((ext_vector_type(8)));
typedef float  v8f     __attribute__((ext_vector_type(8)));
typedef u16    u16x8   __attribute__((ext_vector_type(8)));
typedef u32    u32x4   __attribute__((ext_vector_type(4)));
typedef int    i32x8   __attribute__((ext_vector_type(8)));
typedef int    i32x4   __attribute__((ext_vector_type(4)));

#if defined(__has_builtin)
#if __has_builtin(__builtin_amdgcn_tensor_load_to_lds) && __has_builtin(__builtin_amdgcn_s_wait_tensorcnt)
#define HAVE_TDM 1
#endif
#endif

__device__ __forceinline__ u16 f2bf(float f) {
  unsigned u = __float_as_uint(f);
  u += 0x7FFFu + ((u >> 16) & 1u);   // round-to-nearest-even
  return (u16)(u >> 16);
}

#define BMT 64
#define BNT 128
#define BKT 32
#define LSTRIDE 40   // LDS row stride (bf16): 32 data + 4-DWORD TDM pad -> bank spread

#ifdef HAVE_TDM
// ---------------------------------------------------------------------------
// TDM 2D tile load: global [rows x 32 bf16] (row stride = stride_elems) -> LDS,
// zero-fill rows >= rows_avail, LDS padded 16B per 64B row (stride = 80B).
// D# bitfields per CDNA5 ISA 8.3/8.4 (type=2 image, count=1, data_size=2B,
// pad_enable, pad_interval=3 (16 DW), pad_amount=3 (4 DW)).
// 6-arg builtin form: (g0, g1, g2, g3, g4, cpol) with zero-filled extra groups.
// ---------------------------------------------------------------------------
__device__ __forceinline__ void tdm_load_2d(unsigned lds_off, const u16* g,
                                            int rows_avail, int tile_rows,
                                            int stride_elems)
{
  unsigned long long ga = (unsigned long long)(size_t)g;
  u32x4 g0;
  g0[0] = 1u;                                    // count=1, user mode
  g0[1] = lds_off;                               // lds_addr (bytes)
  g0[2] = (u32)(ga & 0xFFFFFFFFull);             // global_addr[31:0]
  g0[3] = (u32)((ga >> 32) & 0x01FFFFFFull) | (2u << 30);  // addr[56:32] | type=2
  i32x8 g1;
  g1[0] = (int)((1u << 16) | (1u << 20) | (3u << 22) | (3u << 25));
  g1[1] = (int)(32u << 16);                      // tensor_dim0 = 32 elems
  g1[2] = (int)(((u32)rows_avail & 0xFFFFu) << 16); // tensor_dim1 = rows avail
  g1[3] = (int)(32u << 16);                      // tile_dim0 = 32
  g1[4] = (int)((u32)tile_rows & 0xFFFFu);       // tile_dim1 (tile_dim2 = 0)
  g1[5] = (int)(u32)stride_elems;                // tensor_dim0_stride (lo32)
  g1[6] = 0;
  g1[7] = 0;
  i32x4 z4 = {0, 0, 0, 0};
  i32x8 z8 = {0, 0, 0, 0, 0, 0, 0, 0};
  __builtin_amdgcn_tensor_load_to_lds(g0, g1, z4, z4, z8, 0);
}
#endif

// ---------------------------------------------------------------------------
// Generic bf16 WMMA GEMM:  C[M,N] = act(A[M,Kp] * W[N,Kp]^T + bias) + res
// A, W bf16 row-major (Kp multiple of 32, zero-padded). Out f32 (Cf) or bf16 (Cb).
// Batched via blockIdx.z. Block: 256 threads (8 waves), tile 64(M)x128(N), K=32.
// Tiles staged global->LDS by the Tensor Data Mover (one issue per workgroup).
// ---------------------------------------------------------------------------
__global__ __launch_bounds__(256)
void k_gemm(const u16* __restrict__ A, int lda, long long sA,
            const u16* __restrict__ W, int ldw, long long sW,
            const float* __restrict__ bias,
            float* __restrict__ Cf, u16* __restrict__ Cb,
            int ldc, long long sC,
            const float* __restrict__ res,
            int M, int N, int Kp, int act)
{
  __shared__ u16 As[BMT * LSTRIDE];
  __shared__ u16 Bs[BNT * LSTRIDE];
  const int tid  = threadIdx.x;
  const int wave = tid >> 5;
  const int lane = tid & 31;
  const int bm0  = blockIdx.y * BMT;
  const int bn0  = blockIdx.x * BNT;
  const long long z = blockIdx.z;
  A += z * sA;  W += z * sW;
  if (Cf)  Cf  += z * sC;
  if (Cb)  Cb  += z * sC;
  if (res) res += z * sC;

  const int wm = wave & 3;         // m-tile of this wave (0..3)
  const int wn = (wave >> 2) * 4;  // first n-tile (0 or 4)
  const int hl  = lane >> 4;       // lane half
  const int l16 = lane & 15;

  const u16* Atile = A + (long long)bm0 * lda;
  const u16* Wtile = W + (long long)bn0 * ldw;
  const int mRem = M - bm0;
  const int nRem = N - bn0;
#ifdef HAVE_TDM
  const unsigned ldsA = (unsigned)(size_t)(void*)As;
  const unsigned ldsB = (unsigned)(size_t)(void*)Bs;
#endif

  v8f zero8 = {0.f,0.f,0.f,0.f,0.f,0.f,0.f,0.f};
  v8f acc[4] = {zero8, zero8, zero8, zero8};

  for (int k0 = 0; k0 < Kp; k0 += BKT) {
#ifdef HAVE_TDM
    if (tid == 0) {   // TDM ignores EXEC; issue once per workgroup
      tdm_load_2d(ldsA, Atile + k0, mRem, BMT, lda);
      tdm_load_2d(ldsB, Wtile + k0, nRem, BNT, ldw);
      __builtin_amdgcn_s_wait_tensorcnt(0);
    }
    __syncthreads();
#else
    { // manual staging fallback
      int row = tid >> 2;
      int cc  = (tid & 3) * 8;
      u16x8 v = {0,0,0,0,0,0,0,0};
      if (row < mRem) v = *(const u16x8*)(Atile + (long long)row * lda + k0 + cc);
      *(u16x8*)(As + row * LSTRIDE + cc) = v;
    }
    #pragma unroll
    for (int r = 0; r < 2; ++r) {
      int chunk = tid + r * 256;
      int row = chunk >> 2;
      int cc  = (chunk & 3) * 8;
      u16x8 v = {0,0,0,0,0,0,0,0};
      if (row < nRem) v = *(const u16x8*)(Wtile + (long long)row * ldw + k0 + cc);
      *(u16x8*)(Bs + row * LSTRIDE + cc) = v;
    }
    __syncthreads();
#endif

    // A fragment per ISA 16-bit 16x32 layout:
    // lane half 0: K 0..7 & 16..23 ; half 1: K 8..15 & 24..31, row = l16
    bf16x16 af;
    {
      const u16* p = As + (wm * 16 + l16) * LSTRIDE;
      union { u16x8 u; bf16x8 b; } lo, hi;
      lo.u = *(const u16x8*)(p + hl * 8);
      hi.u = *(const u16x8*)(p + 16 + hl * 8);
      #pragma unroll
      for (int i = 0; i < 8; ++i) { af[i] = lo.b[i]; af[i + 8] = hi.b[i]; }
    }
    #pragma unroll
    for (int j = 0; j < 4; ++j) {
      // B fragment: col = l16, lane half 0: K 0..7 & 16..23 ; half 1: K 8..15 & 24..31
      const u16* p = Bs + ((wn + j) * 16 + l16) * LSTRIDE + hl * 8;
      union { u16x8 u; bf16x8 b; } lo, hi;
      lo.u = *(const u16x8*)(p);
      hi.u = *(const u16x8*)(p + 16);
      bf16x16 bf;
      #pragma unroll
      for (int i = 0; i < 8; ++i) { bf[i] = lo.b[i]; bf[i + 8] = hi.b[i]; }
      acc[j] = __builtin_amdgcn_wmma_f32_16x16x32_bf16(
          false, af, false, bf, (short)0, acc[j], false, false);
    }
    __syncthreads();
  }

  // Epilogue. C/D layout: VGPR g -> row g + 8*half, lane l16 -> col.
  #pragma unroll
  for (int j = 0; j < 4; ++j) {
    int n = bn0 + (wn + j) * 16 + l16;
    if (n >= N) continue;
    float bv = bias ? bias[n] : 0.f;
    #pragma unroll
    for (int g = 0; g < 8; ++g) {
      int m = bm0 + wm * 16 + hl * 8 + g;
      if (m >= M) continue;
      float v = acc[j][g] + bv;
      if (act == 1) v = v / (1.f + __expf(-v));   // SiLU
      long long off = (long long)m * ldc + n;
      if (res) v += res[off];
      if (Cf) Cf[off] = v;
      if (Cb) Cb[off] = f2bf(v);
    }
  }
}

// ---------------------------------------------------------------------------
// LayerNorm over D=512 with optional length masking; bf16 output for next GEMM
// ---------------------------------------------------------------------------
__global__ __launch_bounds__(256)
void k_layernorm(const float* __restrict__ x, u16* __restrict__ out,
                 const float* __restrict__ g, const float* __restrict__ b,
                 const int* __restrict__ lens, int T)
{
  __shared__ float red[256];
  const int row = blockIdx.x;
  const int tid = threadIdx.x;
  bool masked = false;
  if (lens) masked = ((row % T) >= lens[row / T]);
  const float* xr = x + (long long)row * 512;
  float v0 = masked ? 0.f : xr[tid];
  float v1 = masked ? 0.f : xr[tid + 256];
  red[tid] = v0 + v1; __syncthreads();
  for (int st = 128; st > 0; st >>= 1) { if (tid < st) red[tid] += red[tid + st]; __syncthreads(); }
  float mu = red[0] * (1.f / 512.f); __syncthreads();
  float d0 = v0 - mu, d1 = v1 - mu;
  red[tid] = d0 * d0 + d1 * d1; __syncthreads();
  for (int st = 128; st > 0; st >>= 1) { if (tid < st) red[tid] += red[tid + st]; __syncthreads(); }
  float rs = rsqrtf(red[0] * (1.f / 512.f) + GEPS);
  long long o = (long long)row * 512;
  out[o + tid]       = f2bf(d0 * rs * g[tid]       + b[tid]);
  out[o + tid + 256] = f2bf(d1 * rs * g[tid + 256] + b[tid + 256]);
}

// softmax over 256 cols (with 1/sqrt(hd) scale), bf16 out
__global__ __launch_bounds__(256)
void k_softmax(const float* __restrict__ s, u16* __restrict__ out, float scale)
{
  __shared__ float red[256];
  const long long row = blockIdx.x;
  const int tid = threadIdx.x;
  float v = s[row * 256 + tid] * scale;
  red[tid] = v; __syncthreads();
  for (int st = 128; st > 0; st >>= 1) { if (tid < st) red[tid] = fmaxf(red[tid], red[tid + st]); __syncthreads(); }
  float mx = red[0]; __syncthreads();
  float e = __expf(v - mx);
  red[tid] = e; __syncthreads();
  for (int st = 128; st > 0; st >>= 1) { if (tid < st) red[tid] += red[tid + st]; __syncthreads(); }
  out[row * 256 + tid] = f2bf(e / red[0]);
}

// batchnorm (inference) in place on [rows,512] f32
__global__ void k_bnorm(float* __restrict__ h, const float* g, const float* b,
                        const float* m, const float* v, int n)
{
  int i = blockIdx.x * 256 + threadIdx.x;
  if (i >= n) return;
  int c = i & 511;
  h[i] = (h[i] - m[c]) * rsqrtf(v[c] + GEPS) * g[c] + b[c];
}

// f32 [rows,K] -> bf16 [rows,Kp] zero-padded
__global__ void k_cvt_pad(const float* __restrict__ src, u16* __restrict__ dst,
                          int rows, int K, int Kp)
{
  int i = blockIdx.x * 256 + threadIdx.x;
  if (i >= rows * Kp) return;
  int r = i / Kp, c = i - r * Kp;
  dst[i] = (c < K) ? f2bf(src[(long long)r * K + c]) : (u16)0;
}

// |spectrogram| -> padded bf16 [B*Tspec, 288] (F=257)
__global__ void k_mag(const float* __restrict__ x, u16* __restrict__ dst)
{
  int i = blockIdx.x * 256 + threadIdx.x;
  if (i >= 4096 * 288) return;
  int r = i / 288, c = i - r * 288;
  u16 o = 0;
  if (c < 257) {
    int b = r >> 10, t = r & 1023;
    float re = x[(( (long long)b * 2 + 0) * 1024 + t) * 257 + c];
    float im = x[(( (long long)b * 2 + 1) * 1024 + t) * 257 + c];
    o = f2bf(sqrtf(re * re + im * im));
  }
  dst[i] = o;
}

// im2col for causal conv: mel f32 [B,1024,80] -> bf16 [B*256, 736] (K=720 pad 736)
__global__ void k_im2col(const float* __restrict__ mel, u16* __restrict__ dst)
{
  int i = blockIdx.x * 256 + threadIdx.x;
  if (i >= 1024 * 736) return;
  int r = i / 736, c = i - r * 736;
  u16 o = 0;
  if (c < 720) {
    int b = r >> 8, t = r & 255;
    int kt = c / 80, m = c - kt * 80;
    int ti = 4 * t + kt - 5;             // left pad 5, stride 4
    if (ti >= 0 && ti < 1024) o = f2bf(mel[((long long)b * 1024 + ti) * 80 + m]);
  }
  dst[i] = o;
}

// conv_w [512,80,9] -> im2col-ordered bf16 [512,736]
__global__ void k_convw(const float* __restrict__ w, u16* __restrict__ dst)
{
  int i = blockIdx.x * 256 + threadIdx.x;
  if (i >= 512 * 736) return;
  int d = i / 736, c = i - d * 736;
  u16 o = 0;
  if (c < 720) {
    int kt = c / 80, m = c - kt * 80;
    o = f2bf(w[((long long)d * 80 + m) * 9 + kt]);
  }
  dst[i] = o;
}

// q/k f32 [B*T,512] -> bf16 [B,H,T,64]
__global__ void k_qk_split(const float* __restrict__ src, u16* __restrict__ dst)
{
  int i = blockIdx.x * 256 + threadIdx.x;
  if (i >= 524288) return;
  int d = i & 63, h = (i >> 6) & 7, t = (i >> 9) & 255, b = i >> 17;
  dst[(((long long)(b * 8 + h) * 256 + t) * 64) + d] = f2bf(src[i]);
}
// v f32 [B*T,512] -> bf16 [B,H,64,T] (transposed for A*V^T form)
__global__ void k_vT(const float* __restrict__ src, u16* __restrict__ dst)
{
  int i = blockIdx.x * 256 + threadIdx.x;
  if (i >= 524288) return;
  int d = i & 63, h = (i >> 6) & 7, t = (i >> 9) & 255, b = i >> 17;
  dst[(((long long)(b * 8 + h) * 64 + d) * 256) + t] = f2bf(src[i]);
}
// o f32 [B,H,T,64] -> bf16 [B*T,512]
__global__ void k_omerge(const float* __restrict__ src, u16* __restrict__ dst)
{
  int i = blockIdx.x * 256 + threadIdx.x;
  if (i >= 524288) return;
  int d = i & 63, t = (i >> 6) & 255, h = (i >> 14) & 7, b = i >> 17;
  dst[((long long)(b * 256 + t) * 512) + h * 64 + d] = f2bf(src[i]);
}

// embedding lookup -> bf16 [B*U,512]
__global__ void k_embed(const int* __restrict__ tok, const float* __restrict__ emb,
                        u16* __restrict__ dst)
{
  int i = blockIdx.x * 256 + threadIdx.x;
  if (i >= 262144) return;
  int r = i >> 9, c = i & 511;
  dst[i] = f2bf(emb[(long long)tok[r] * 512 + c]);
}

// ---------------------------------------------------------------------------
// Persistent LSTM recurrence: one block per batch element.
// xg = x@Wih^T + bih precomputed via WMMA GEMM.  Whh kept f32 (L2-resident).
// 1024 threads: thread j dots Whh rows j and j+1024 against h (LDS).
// ---------------------------------------------------------------------------
__global__ __launch_bounds__(1024)
void k_lstm(const float* __restrict__ xg, const float* __restrict__ Whh,
            const float* __restrict__ bhh, float* __restrict__ hs, int T)
{
  __shared__ float hsh[512];
  __shared__ float csh[512];
  __shared__ float gbuf[2048];
  const int b = blockIdx.x;
  const int tid = threadIdx.x;
  if (tid < 512) { hsh[tid] = 0.f; csh[tid] = 0.f; }
  __syncthreads();
  const float4* w0 = (const float4*)(Whh + (long long)tid * 512);
  const float4* w1 = (const float4*)(Whh + (long long)(tid + 1024) * 512);
  for (int t = 0; t < T; ++t) {
    float a0 = 0.f, a1 = 0.f;
    const float4* hp = (const float4*)hsh;
    #pragma unroll 4
    for (int k = 0; k < 128; ++k) {
      float4 h4 = hp[k];
      float4 x0 = w0[k], x1 = w1[k];
      a0 += h4.x * x0.x + h4.y * x0.y + h4.z * x0.z + h4.w * x0.w;
      a1 += h4.x * x1.x + h4.y * x1.y + h4.z * x1.z + h4.w * x1.w;
    }
    gbuf[tid] = a0; gbuf[tid + 1024] = a1;
    __syncthreads();
    if (tid < 512) {
      const float* xr = xg + ((long long)(b * T + t)) * 2048;
      float iv = gbuf[tid]        + xr[tid]        + bhh[tid];
      float fv = gbuf[tid + 512]  + xr[tid + 512]  + bhh[tid + 512];
      float gv = gbuf[tid + 1024] + xr[tid + 1024] + bhh[tid + 1024];
      float ov = gbuf[tid + 1536] + xr[tid + 1536] + bhh[tid + 1536];
      float si = 1.f / (1.f + __expf(-iv));
      float sf = 1.f / (1.f + __expf(-fv));
      float so = 1.f / (1.f + __expf(-ov));
      float c  = sf * csh[tid] + si * tanhf(gv);
      float h  = so * tanhf(c);
      csh[tid] = c; hsh[tid] = h;
      hs[((long long)(b * T + t)) * 512 + tid] = h;
    }
    __syncthreads();
  }
}

// rnnt_logits[b,t,u,:] = Apart[b,t,:] + Bpart[b,u,:]  (537 MB streaming stores)
__global__ __launch_bounds__(256)
void k_bcast(const float* __restrict__ Ap, const float* __restrict__ Bp,
             float* __restrict__ out)
{
  long long blk = blockIdx.x;             // B*T*U blocks
  int u = (int)(blk & 127);
  long long bt = blk >> 7;                // b*T + t
  const float4* ar = (const float4*)(Ap + bt * 1024);
  const float4* br = (const float4*)(Bp + (((bt >> 8) * 128) + u) * 1024);
  float4* o = (float4*)(out + blk * 1024);
  int i = threadIdx.x;                    // 256 * float4 = 1024
  float4 a = ar[i], c = br[i];
  float4 r; r.x = a.x + c.x; r.y = a.y + c.y; r.z = a.z + c.z; r.w = a.w + c.w;
  o[i] = r;
}

// ===========================================================================
extern "C" void kernel_launch(void* const* d_in, const int* in_sizes, int n_in,
                              void* d_out, int out_size, void* d_ws, size_t ws_size,
                              hipStream_t stream) {
  (void)in_sizes; (void)n_in; (void)out_size; (void)ws_size;
  const float* IX   = (const float*)d_in[0];
  const int* audio_len   = (const int*)d_in[1];
  const int* targets     = (const int*)d_in[2];
  const int* targets_len = (const int*)d_in[3];
  auto P = [&](int i) { return (const float*)d_in[i]; };
  // param indices (setup_inputs dict order)
  const int I_MELW=4, I_CONVW=5, I_CONVB=6, I_BNG=7, I_BNB=8, I_BNM=9, I_BNV=10;
  const int I_L0=11; // 16 leaves per layer * 4 layers -> 11..74
  const int I_PNG=75, I_PNB=76, I_CWIH=77, I_CWHH=78, I_CBIH=79, I_CBHH=80;
  const int I_CNG=81, I_CNB=82, I_CC1=83, I_CB1=84, I_CC2=85, I_CB2=86, I_EMB=87;
  const int I_RWIH=88, I_RWHH=89, I_RBIH=90, I_RBHH=91, I_RNG=92, I_RNB=93;
  const int I_JEW=94, I_JEB=95, I_JPW=96, I_JPB=97, I_JOW=98, I_JOB=99;

  char* ws = (char*)d_ws;
  size_t off = 0;
  auto alloc = [&](size_t bytes) -> void* {
    void* p = ws + off; off += (bytes + 255) & ~(size_t)255; return p;
  };
  // scratch
  u16*   magp  = (u16*)  alloc(4096 * 288 * 2);
  u16*   melw  = (u16*)  alloc(80 * 288 * 2);
  float* melf  = (float*)alloc(4096 * 80 * 4);
  u16*   im2   = (u16*)  alloc(1024 * 736 * 2);
  u16*   convw = (u16*)  alloc(512 * 736 * 2);
  float* xbuf  = (float*)alloc(1024 * 512 * 4);
  u16*   lnbuf = (u16*)  alloc(1024 * 512 * 2);
  float* qf    = (float*)alloc(1024 * 512 * 4);
  float* kf    = (float*)alloc(1024 * 512 * 4);
  float* vf    = (float*)alloc(1024 * 512 * 4);
  u16*   qh    = (u16*)  alloc(524288 * 2);
  u16*   kh    = (u16*)  alloc(524288 * 2);
  u16*   vt    = (u16*)  alloc(524288 * 2);
  float* scores= (float*)alloc((size_t)32 * 65536 * 4);
  u16*   attn  = (u16*)  alloc((size_t)32 * 65536 * 2);
  float* oh    = (float*)alloc(524288 * 4);
  u16*   obf   = (u16*)  alloc(524288 * 2);
  u16*   ffh   = (u16*)  alloc(1024 * 2048 * 2);
  u16*   lw[4][6]; // wq wk wv wo w1 w2 per layer (bf16)
  for (int l = 0; l < 4; ++l) {
    lw[l][0] = (u16*)alloc(512 * 512 * 2);
    lw[l][1] = (u16*)alloc(512 * 512 * 2);
    lw[l][2] = (u16*)alloc(512 * 512 * 2);
    lw[l][3] = (u16*)alloc(512 * 512 * 2);
    lw[l][4] = (u16*)alloc(2048 * 512 * 2);
    lw[l][5] = (u16*)alloc(512 * 2048 * 2);
  }
  u16*   encln = (u16*)  alloc(1024 * 512 * 2);
  u16*   cwih  = (u16*)  alloc(2048 * 512 * 2);
  float* xg    = (float*)alloc(1024 * 2048 * 4);
  float* hsb   = (float*)alloc(1024 * 512 * 4);
  u16*   cbf   = (u16*)  alloc(1024 * 512 * 2);
  u16*   cw1   = (u16*)  alloc(1024 * 512 * 2);
  u16*   h1    = (u16*)  alloc(1024 * 1024 * 2);
  u16*   cw2   = (u16*)  alloc(1024 * 1024 * 2);
  u16*   embbf = (u16*)  alloc(512 * 512 * 2);
  u16*   rwih  = (u16*)  alloc(2048 * 512 * 2);
  float* xgr   = (float*)alloc(512 * 2048 * 4);
  float* hsr   = (float*)alloc(512 * 512 * 4);
  u16*   dbf   = (u16*)  alloc(512 * 512 * 2);
  u16*   jebf  = (u16*)  alloc(768 * 512 * 2);
  u16*   jpbf  = (u16*)  alloc(768 * 512 * 2);
  u16*   jobf  = (u16*)  alloc(1024 * 1536 * 2);
  u16*   encj  = (u16*)  alloc(1024 * 768 * 2);
  u16*   predj = (u16*)  alloc(512 * 768 * 2);
  float* Apart = (float*)alloc(1024 * 1024 * 4);
  float* Bpart = (float*)alloc(512 * 1024 * 4);

  auto ew = [](int n) { return dim3((n + 255) / 256); };
  auto cvt = [&](const float* s, u16* d, int rows, int K, int Kp) {
    k_cvt_pad<<<ew(rows * Kp), 256, 0, stream>>>(s, d, rows, K, Kp);
  };
  auto gemm = [&](const u16* A, int lda, long long sA, const u16* W, int ldw, long long sW,
                  const float* bias, float* Cf, u16* Cb, int ldc, long long sC,
                  const float* res, int M, int N, int Kp, int act, int batch) {
    dim3 g((N + BNT - 1) / BNT, (M + BMT - 1) / BMT, batch);
    k_gemm<<<g, 256, 0, stream>>>(A, lda, sA, W, ldw, sW, bias, Cf, Cb, ldc, sC, res, M, N, Kp, act);
  };

  // ---- frontend: |spec| -> mel -> causal conv (as GEMM) -> BN ----
  k_mag<<<ew(4096 * 288), 256, 0, stream>>>(IX, magp);
  cvt(P(I_MELW), melw, 80, 257, 288);
  gemm(magp, 288, 0, melw, 288, 0, nullptr, melf, nullptr, 80, 0, nullptr, 4096, 80, 288, 0, 1);
  k_im2col<<<ew(1024 * 736), 256, 0, stream>>>(melf, im2);
  k_convw<<<ew(512 * 736), 256, 0, stream>>>(P(I_CONVW), convw);
  gemm(im2, 736, 0, convw, 736, 0, P(I_CONVB), xbuf, nullptr, 512, 0, nullptr, 1024, 512, 736, 0, 1);
  k_bnorm<<<ew(1024 * 512), 256, 0, stream>>>(xbuf, P(I_BNG), P(I_BNB), P(I_BNM), P(I_BNV), 1024 * 512);

  // ---- convert all GEMM weights to bf16 once per call ----
  for (int l = 0; l < 4; ++l) {
    int base = I_L0 + l * 16;
    cvt(P(base + 2),  lw[l][0], 512, 512, 512);   // wq
    cvt(P(base + 4),  lw[l][1], 512, 512, 512);   // wk
    cvt(P(base + 6),  lw[l][2], 512, 512, 512);   // wv
    cvt(P(base + 8),  lw[l][3], 512, 512, 512);   // wo
    cvt(P(base + 12), lw[l][4], 2048, 512, 512);  // w1
    cvt(P(base + 14), lw[l][5], 512, 2048, 2048); // w2
  }
  cvt(P(I_CWIH), cwih, 2048, 512, 512);
  cvt(P(I_CC1),  cw1,  1024, 512, 512);
  cvt(P(I_CC2),  cw2,  1024, 1024, 1024);
  cvt(P(I_RWIH), rwih, 2048, 512, 512);
  cvt(P(I_JEW),  jebf, 768, 512, 512);
  cvt(P(I_JPW),  jpbf, 768, 512, 512);
  cvt(P(I_JOW),  jobf, 1024, 1536, 1536);

  // ---- encoder: 4 x (pre-LN MHSA + SiLU FFN), all GEMMs on WMMA ----
  for (int l = 0; l < 4; ++l) {
    int base = I_L0 + l * 16;
    k_layernorm<<<1024, 256, 0, stream>>>(xbuf, lnbuf, P(base + 0), P(base + 1), nullptr, 256);
    gemm(lnbuf, 512, 0, lw[l][0], 512, 0, P(base + 3), qf, nullptr, 512, 0, nullptr, 1024, 512, 512, 0, 1);
    gemm(lnbuf, 512, 0, lw[l][1], 512, 0, P(base + 5), kf, nullptr, 512, 0, nullptr, 1024, 512, 512, 0, 1);
    gemm(lnbuf, 512, 0, lw[l][2], 512, 0, P(base + 7), vf, nullptr, 512, 0, nullptr, 1024, 512, 512, 0, 1);
    k_qk_split<<<ew(524288), 256, 0, stream>>>(qf, qh);
    k_qk_split<<<ew(524288), 256, 0, stream>>>(kf, kh);
    k_vT<<<ew(524288), 256, 0, stream>>>(vf, vt);
    // scores[b,h] = Q[256,64] * K[256,64]^T   (batch = 32)
    gemm(qh, 64, 16384, kh, 64, 16384, nullptr, scores, nullptr, 256, 65536, nullptr, 256, 256, 64, 0, 32);
    k_softmax<<<8192, 256, 0, stream>>>(scores, attn, 0.125f);
    // o[b,h] = A[256,256] * V^T[64,256]^T
    gemm(attn, 256, 65536, vt, 256, 16384, nullptr, oh, nullptr, 64, 16384, nullptr, 256, 64, 256, 0, 32);
    k_omerge<<<ew(524288), 256, 0, stream>>>(oh, obf);
    gemm(obf, 512, 0, lw[l][3], 512, 0, P(base + 9), xbuf, nullptr, 512, 0, xbuf, 1024, 512, 512, 0, 1);
    k_layernorm<<<1024, 256, 0, stream>>>(xbuf, lnbuf, P(base + 10), P(base + 11), nullptr, 256);
    gemm(lnbuf, 512, 0, lw[l][4], 512, 0, P(base + 13), nullptr, ffh, 2048, 0, nullptr, 1024, 2048, 512, 1, 1);
    gemm(ffh, 2048, 0, lw[l][5], 2048, 0, P(base + 15), xbuf, nullptr, 512, 0, xbuf, 1024, 512, 2048, 0, 1);
  }
  k_layernorm<<<1024, 256, 0, stream>>>(xbuf, encln, P(I_PNG), P(I_PNB), nullptr, 256);

  // ---- CTC branch ----
  gemm(encln, 512, 0, cwih, 512, 0, P(I_CBIH), xg, nullptr, 2048, 0, nullptr, 1024, 2048, 512, 0, 1);
  k_lstm<<<4, 1024, 0, stream>>>(xg, P(I_CWHH), P(I_CBHH), hsb, 256);
  k_layernorm<<<1024, 256, 0, stream>>>(hsb, cbf, P(I_CNG), P(I_CNB), audio_len, 256);
  gemm(cbf, 512, 0, cw1, 512, 0, P(I_CB1), nullptr, h1, 1024, 0, nullptr, 1024, 1024, 512, 1, 1);
  gemm(h1, 1024, 0, cw2, 1024, 0, P(I_CB2), (float*)d_out, nullptr, 1024, 0, nullptr, 1024, 1024, 1024, 0, 1);

  // ---- RNNT prediction network ----
  k_embed<<<ew(262144), 256, 0, stream>>>(targets, P(I_EMB), embbf);
  gemm(embbf, 512, 0, rwih, 512, 0, P(I_RBIH), xgr, nullptr, 2048, 0, nullptr, 512, 2048, 512, 0, 1);
  k_lstm<<<4, 1024, 0, stream>>>(xgr, P(I_RWHH), P(I_RBHH), hsr, 128);
  k_layernorm<<<512, 256, 0, stream>>>(hsr, dbf, P(I_RNG), P(I_RNB), targets_len, 128);

  // ---- joint: enc_j/pred_j, split jo_w GEMMs, then broadcast add ----
  gemm(encln, 512, 0, jebf, 512, 0, P(I_JEB), nullptr, encj, 768, 0, nullptr, 1024, 768, 512, 0, 1);
  gemm(dbf, 512, 0, jpbf, 512, 0, P(I_JPB), nullptr, predj, 768, 0, nullptr, 512, 768, 512, 0, 1);
  gemm(encj, 768, 0, jobf, 1536, 0, P(I_JOB), Apart, nullptr, 1024, 0, nullptr, 1024, 1024, 768, 0, 1);
  gemm(predj, 768, 0, jobf + 768, 1536, 0, nullptr, Bpart, nullptr, 1024, 0, nullptr, 512, 1024, 768, 0, 1);
  k_bcast<<<dim3(131072), 256, 0, stream>>>(Apart, Bpart, (float*)d_out + 1048576);
}
